// CausalSelfAttention_3289944949411
// MI455X (gfx1250) — matrix operand order repaired and checked by reading, not compiled, not run on hardware
//
#include <hip/hip_runtime.h>

#define TSEQ 2048
#define CDIM 2048
#define NHEAD 32
#define NKV 8
#define HDIM 64
#define RTOT (2 * TSEQ)   // B*T total rows

typedef __attribute__((ext_vector_type(16))) __bf16 v16bf;
typedef __attribute__((ext_vector_type(8)))  __bf16 v8bf;
typedef __attribute__((ext_vector_type(8)))  float  v8f;
typedef __attribute__((ext_vector_type(4)))  int    v4i;

typedef const __attribute__((address_space(1))) void* gas_vp;
typedef __attribute__((address_space(3))) void*       las_vp;
typedef __attribute__((address_space(1))) v4i*        gas_v4;
typedef __attribute__((address_space(3))) v4i*        las_v4;

#if defined(__has_builtin)
#  if __has_builtin(__builtin_amdgcn_global_load_async_to_lds_b128)
#    define ASYNC_LDS 1
#  endif
#  if __has_builtin(__builtin_amdgcn_s_wait_asynccnt)
#    define HAVE_WAIT_ASYNC 1
#  endif
#endif

#if defined(ASYNC_LDS)
// b128 async copy: global -> LDS, ASYNCcnt-tracked, no VGPR round trip.
static __device__ __forceinline__ void async_ld128(const void* g, void* l) {
  __builtin_amdgcn_global_load_async_to_lds_b128(
      (gas_v4)(gas_vp)g, (las_v4)(las_vp)l, 0, 0);
}
#endif

static __device__ __forceinline__ void wait_async0() {
#if defined(HAVE_WAIT_ASYNC)
  __builtin_amdgcn_s_wait_asynccnt(0);
#elif defined(ASYNC_LDS)
  asm volatile("s_wait_asynccnt 0x0" ::: "memory");
#endif
}

static __device__ __forceinline__ __bf16 f2bf(float f) {
  union { float f; unsigned u; } v; v.f = f;
  unsigned r = v.u + 0x7FFFu + ((v.u >> 16) & 1u);   // round-to-nearest-even
  unsigned short h = (unsigned short)(r >> 16);
  return __builtin_bit_cast(__bf16, h);
}
static __device__ __forceinline__ float bf2f(__bf16 b) {
  unsigned short h = __builtin_bit_cast(unsigned short, b);
  union { unsigned u; float f; } v; v.u = ((unsigned)h) << 16;
  return v.f;
}
static __device__ __forceinline__ v8f vzero8() {
  v8f z;
  #pragma unroll
  for (int e = 0; e < 8; ++e) z[e] = 0.0f;
  return z;
}
static __device__ __forceinline__ v16bf combine(v8bf lo, v8bf hi) {
  v16bf r;
  #pragma unroll
  for (int e = 0; e < 8; ++e) { r[e] = lo[e]; r[e + 8] = hi[e]; }
  return r;
}
static __device__ __forceinline__ void stc(float* p, float v) { *p = v; }
static __device__ __forceinline__ void stc(__bf16* p, float v) { *p = f2bf(v); }

// ---------------------------------------------------------------- convert f32 -> bf16
__global__ void cvt_f32_bf16(const float* __restrict__ src, __bf16* __restrict__ dst, int n) {
  int i = blockIdx.x * blockDim.x + threadIdx.x;
  if (i < n) dst[i] = f2bf(src[i]);
}
// convert + transpose: src f32 [K,N] row-major -> dst bf16 [N,K] row-major
__global__ void cvt_t_f32_bf16(const float* __restrict__ src, __bf16* __restrict__ dst,
                               int K, int N) {
  int idx = blockIdx.x * blockDim.x + threadIdx.x;
  if (idx >= K * N) return;
  int k = idx % K, n = idx / K;            // dst-major ordering: writes coalesced
  dst[(size_t)n * K + k] = f2bf(src[(size_t)k * N + n]);
}

// ---------------------------------------------------------------- bf16 WMMA GEMM
// C = A @ B with A [M,K] row-major, BT [N,K] row-major (B pre-transposed).
// TRC=false: C[row*N+col]; TRC=true: C[col*M+row] (transposed output for V).
// Tile 128x128x64, double-buffered LDS, all staging async (direct row copies).
// A frag (16-bit 16x32): lane<16 -> K 0-7 & 16-23; lane>=16 -> K 8-15 & 24-31 (2x b128).
// B frag (16-bit 32x16): lane = half*16+n, same K order as A frag (2x b128).
template <typename OUT, bool TRC>
__global__ __launch_bounds__(256, 1)
void gemm_bf16_wmma(const __bf16* __restrict__ A, const __bf16* __restrict__ BT,
                    OUT* __restrict__ C, int M, int N, int K) {
  constexpr int BK = 64, LDT = 72;   // pad: 144B rows stay 16B aligned
  __shared__ __bf16 As[2][128 * LDT];
  __shared__ __bf16 Bs[2][128 * LDT];   // Bs[n][k]
  const int tid  = threadIdx.x;
  const int lane = tid & 31;
  const int wave = tid >> 5;
  const int hi16 = lane >> 4;
  const int l16  = lane & 15;
  const int wm = (wave & 3) * 32;
  const int wn = (wave >> 2) * 64;
  const int m0 = blockIdx.y * 128;
  const int n0 = blockIdx.x * 128;

  v8f acc[2][4];
  #pragma unroll
  for (int i = 0; i < 2; ++i)
    #pragma unroll
    for (int j = 0; j < 4; ++j) acc[i][j] = vzero8();

  const int s_chunk = tid & 7;   // 8 chunks of 8 bf16 across K=64
  const int s_row   = tid >> 3;  // 0..31, 4 passes of 32 rows

  auto stageA = [&](int buf, int kb) {
#if defined(ASYNC_LDS)
    #pragma unroll
    for (int p = 0; p < 4; ++p) {
      int r = s_row + p * 32;
      async_ld128(A + (size_t)(m0 + r) * K + kb + s_chunk * 8,
                  &As[buf][r * LDT + s_chunk * 8]);
    }
#else
    v8bf t[4];
    #pragma unroll
    for (int p = 0; p < 4; ++p)
      t[p] = *(const v8bf*)(A + (size_t)(m0 + s_row + p * 32) * K + kb + s_chunk * 8);
    #pragma unroll
    for (int p = 0; p < 4; ++p)
      *(v8bf*)(&As[buf][(s_row + p * 32) * LDT + s_chunk * 8]) = t[p];
#endif
  };
  auto stageB = [&](int buf, int kb) {   // BT rows are K-contiguous: direct copy
#if defined(ASYNC_LDS)
    #pragma unroll
    for (int p = 0; p < 4; ++p) {
      int r = s_row + p * 32;
      async_ld128(BT + (size_t)(n0 + r) * K + kb + s_chunk * 8,
                  &Bs[buf][r * LDT + s_chunk * 8]);
    }
#else
    v8bf t[4];
    #pragma unroll
    for (int p = 0; p < 4; ++p)
      t[p] = *(const v8bf*)(BT + (size_t)(n0 + s_row + p * 32) * K + kb + s_chunk * 8);
    #pragma unroll
    for (int p = 0; p < 4; ++p)
      *(v8bf*)(&Bs[buf][(s_row + p * 32) * LDT + s_chunk * 8]) = t[p];
#endif
  };

  stageA(0, 0);
  stageB(0, 0);
  wait_async0();
  __syncthreads();

  const int niter = K / BK;
  for (int it = 0; it < niter; ++it) {
    const int cur = it & 1, nxt = cur ^ 1;
    if (it + 1 < niter) {               // prefetch next tile while computing
      stageA(nxt, (it + 1) * BK);
      stageB(nxt, (it + 1) * BK);
    }
    #pragma unroll
    for (int kk = 0; kk < BK; kk += 32) {
      v16bf af[2], bfr[4];
      #pragma unroll
      for (int i = 0; i < 2; ++i) {
        const __bf16* p = &As[cur][(wm + i * 16 + l16) * LDT + kk + hi16 * 8];
        af[i] = combine(*(const v8bf*)p, *(const v8bf*)(p + 16));
      }
      #pragma unroll
      for (int j = 0; j < 4; ++j) {
        const __bf16* p = &Bs[cur][(wn + j * 16 + l16) * LDT + kk + hi16 * 8];
        bfr[j] = combine(*(const v8bf*)p, *(const v8bf*)(p + 16));
      }
      #pragma unroll
      for (int j = 0; j < 4; ++j)
        #pragma unroll
        for (int i = 0; i < 2; ++i)
          acc[i][j] = __builtin_amdgcn_wmma_f32_16x16x32_bf16(
              false, af[i], false, bfr[j], (short)0, acc[i][j], false, false);
    }
    wait_async0();    // next-tile async LDS writes complete before barrier
    __syncthreads();
  }
  // C/D layout: lane holds col = l16, rows r + 8*hi16
  #pragma unroll
  for (int i = 0; i < 2; ++i)
    #pragma unroll
    for (int j = 0; j < 4; ++j) {
      size_t row0 = (size_t)m0 + wm + i * 16 + 8 * hi16;
      size_t col  = (size_t)n0 + wn + j * 16 + l16;
      #pragma unroll
      for (int r = 0; r < 8; ++r) {
        if constexpr (TRC)
          stc(C + col * (size_t)M + row0 + r, acc[i][j][r]);
        else
          stc(C + (row0 + r) * N + col, acc[i][j][r]);
      }
    }
}

// ---------------------------------------------------------------- RoPE (in place, bf16)
__global__ void rope_bf16(__bf16* __restrict__ t, int rows, int heads, int stride) {
  int idx = blockIdx.x * blockDim.x + threadIdx.x;
  if (idx >= rows * heads * (HDIM / 2)) return;
  int i    = idx & 31;
  int tmp  = idx >> 5;
  int head = tmp % heads;
  int row  = tmp / heads;
  int pos  = row & (TSEQ - 1);                       // row = b*T + t
  // inv_freq = 500000^(-2i/64) = exp2(-(2i/64)*log2(500000))
  float ang = (float)pos * exp2f(-(float)(2 * i) * (18.93156856932417f / 64.0f));
  float s, c;
  sincosf(ang, &s, &c);
  __bf16* p = t + (size_t)row * stride + head * HDIM;
  float x0 = bf2f(p[i]);
  float x1 = bf2f(p[i + 32]);
  p[i]      = f2bf(x0 * c - x1 * s);
  p[i + 32] = f2bf(x1 * c + x0 * s);
}

// ---------------------------------------------------------------- flash attention (causal, GQA)
// grid (T/64, H, B), 128 threads = 4 waves; wave owns 16 query rows.
// q: [b*T+t][h*64+d]; k: [b*T+t][kv*64+d]; vT: [kv*64+d][b*T+t]  (all bf16)
__global__ __launch_bounds__(128, 1)
void flash_attn_bf16(const __bf16* __restrict__ qb, const __bf16* __restrict__ kb,
                     const __bf16* __restrict__ vt, __bf16* __restrict__ yb) {
  constexpr int BR = 64, BC = 64, LDT = 72;
  __shared__ __bf16 Qs[BR * LDT];
  __shared__ __bf16 Ks[BC * LDT];     // Ks[token][d]
  __shared__ __bf16 Vt[HDIM * LDT];   // Vt[d][token]
  __shared__ __bf16 Ps[BR * LDT];     // probabilities (wave-private rows)
  const int tid  = threadIdx.x;
  const int lane = tid & 31;
  const int wave = tid >> 5;
  const int hi16 = lane >> 4;
  const int l16  = lane & 15;
  const int q0 = blockIdx.x * BR;
  const int h  = blockIdx.y;
  const int b  = blockIdx.z;
  const int kv = h >> 2;              // n_rep = 4
  const int wm = wave * 16;

  const __bf16* qbase = qb + (size_t)b * TSEQ * CDIM + (size_t)h * HDIM;
  const __bf16* kbase = kb + (size_t)b * TSEQ * (NKV * HDIM) + (size_t)kv * HDIM;
  const __bf16* vtb   = vt + (size_t)kv * HDIM * RTOT + (size_t)b * TSEQ;  // + d*RTOT + t

  const int chunk = tid & 7;   // 8 chunks of 8 bf16 across 64
  const int lrow  = tid >> 3;  // 0..15

  // --- stage Q tile (async direct copy)
#if defined(ASYNC_LDS)
  #pragma unroll
  for (int p = 0; p < 4; ++p) {
    int r = lrow + p * 16;
    async_ld128(qbase + (size_t)(q0 + r) * CDIM + chunk * 8,
                Qs + r * LDT + chunk * 8);
  }
  wait_async0();
#else
  {
    v8bf t[4];
    #pragma unroll
    for (int p = 0; p < 4; ++p)
      t[p] = *(const v8bf*)(qbase + (size_t)(q0 + lrow + p * 16) * CDIM + chunk * 8);
    #pragma unroll
    for (int p = 0; p < 4; ++p)
      *(v8bf*)(Qs + (lrow + p * 16) * LDT + chunk * 8) = t[p];
  }
#endif
  __syncthreads();

  float m_i[8], l_i[8];
  v8f o[4];
  #pragma unroll
  for (int r = 0; r < 8; ++r) { m_i[r] = -3.0e38f; l_i[r] = 0.0f; }
  #pragma unroll
  for (int j = 0; j < 4; ++j) o[j] = vzero8();

  const float scale = 0.125f;          // 1/sqrt(64)
  const float L2E   = 1.44269504f;

  for (int k0 = 0; k0 <= q0; k0 += BC) {
    // --- stage K tile [token][d] and V tile [d][token]: both direct rows -> async
#if defined(ASYNC_LDS)
    #pragma unroll
    for (int p = 0; p < 4; ++p) {
      int r = lrow + p * 16;
      async_ld128(kbase + (size_t)(k0 + r) * (NKV * HDIM) + chunk * 8,
                  Ks + r * LDT + chunk * 8);
      async_ld128(vtb + (size_t)r * RTOT + k0 + chunk * 8,
                  Vt + r * LDT + chunk * 8);
    }
#else
    {
      v8bf tk[4], tv[4];
      #pragma unroll
      for (int p = 0; p < 4; ++p) {
        tk[p] = *(const v8bf*)(kbase + (size_t)(k0 + lrow + p * 16) * (NKV * HDIM) + chunk * 8);
        tv[p] = *(const v8bf*)(vtb + (size_t)(lrow + p * 16) * RTOT + k0 + chunk * 8);
      }
      #pragma unroll
      for (int p = 0; p < 4; ++p) {
        *(v8bf*)(Ks + (lrow + p * 16) * LDT + chunk * 8) = tk[p];
        *(v8bf*)(Vt + (lrow + p * 16) * LDT + chunk * 8) = tv[p];
      }
    }
#endif
    wait_async0();
    __syncthreads();

    // S = Q @ K^T  (contraction over d; K rows are d-contiguous -> B frag is contiguous)
    v8f s[4];
    #pragma unroll
    for (int j = 0; j < 4; ++j) s[j] = vzero8();
    #pragma unroll
    for (int kk = 0; kk < HDIM; kk += 32) {
      const __bf16* pa = Qs + (wm + l16) * LDT + kk + hi16 * 8;
      v16bf aq = combine(*(const v8bf*)pa, *(const v8bf*)(pa + 16));
      v16bf bk[4];
      #pragma unroll
      for (int j = 0; j < 4; ++j) {
        const __bf16* pb = Ks + (j * 16 + l16) * LDT + kk + hi16 * 8;
        bk[j] = combine(*(const v8bf*)pb, *(const v8bf*)(pb + 16));
      }
      #pragma unroll
      for (int j = 0; j < 4; ++j)
        s[j] = __builtin_amdgcn_wmma_f32_16x16x32_bf16(false, aq, false, bk[j],
                                                       (short)0, s[j], false, false);
    }

    // online softmax (rows live in 16-lane halves; reduce over l16 with shfl_xor)
    float corr[8];
    #pragma unroll
    for (int r = 0; r < 8; ++r) {
      int qrow = q0 + wm + r + 8 * hi16;
      float mx = -3.0e38f;
      #pragma unroll
      for (int j = 0; j < 4; ++j) {
        float v = s[j][r] * scale;
        int col = k0 + j * 16 + l16;
        if (col > qrow) v = -3.0e38f;     // causal mask
        s[j][r] = v;
        mx = fmaxf(mx, v);
      }
      #pragma unroll
      for (int msk = 8; msk >= 1; msk >>= 1) mx = fmaxf(mx, __shfl_xor(mx, msk, 32));
      float mn = fmaxf(m_i[r], mx);
      corr[r] = exp2f((m_i[r] - mn) * L2E);
      m_i[r] = mn;
      float sum = 0.0f;
      #pragma unroll
      for (int j = 0; j < 4; ++j) {
        float pv = exp2f((s[j][r] - mn) * L2E);
        s[j][r] = pv;
        sum += pv;
      }
      #pragma unroll
      for (int msk = 8; msk >= 1; msk >>= 1) sum += __shfl_xor(sum, msk, 32);
      l_i[r] = l_i[r] * corr[r] + sum;
    }
    #pragma unroll
    for (int j = 0; j < 4; ++j)
      #pragma unroll
      for (int r = 0; r < 8; ++r) o[j][r] *= corr[r];

    // C-tile -> A-frag relayout via LDS (wave-private region, no barrier needed)
    #pragma unroll
    for (int j = 0; j < 4; ++j)
      #pragma unroll
      for (int r = 0; r < 8; ++r)
        Ps[(wm + r + 8 * hi16) * LDT + j * 16 + l16] = f2bf(s[j][r]);

    // O += P @ V
    #pragma unroll
    for (int kk = 0; kk < BC; kk += 32) {
      const __bf16* pa = Ps + (wm + l16) * LDT + kk + hi16 * 8;
      v16bf ap = combine(*(const v8bf*)pa, *(const v8bf*)(pa + 16));
      v16bf bv[4];
      #pragma unroll
      for (int jd = 0; jd < 4; ++jd) {
        const __bf16* pb = Vt + (jd * 16 + l16) * LDT + kk + hi16 * 8;
        bv[jd] = combine(*(const v8bf*)pb, *(const v8bf*)(pb + 16));
      }
      #pragma unroll
      for (int jd = 0; jd < 4; ++jd)
        o[jd] = __builtin_amdgcn_wmma_f32_16x16x32_bf16(false, ap, false, bv[jd],
                                                        (short)0, o[jd], false, false);
    }
    __syncthreads();   // before Ks/Vt are overwritten
  }

  __bf16* ybase = yb + (size_t)b * TSEQ * CDIM + (size_t)h * HDIM;
  #pragma unroll
  for (int jd = 0; jd < 4; ++jd)
    #pragma unroll
    for (int r = 0; r < 8; ++r) {
      float val = o[jd][r] / l_i[r];
      ybase[(size_t)(q0 + wm + r + 8 * hi16) * CDIM + jd * 16 + l16] = f2bf(val);
    }
}

// ---------------------------------------------------------------- launcher
extern "C" void kernel_launch(void* const* d_in, const int* in_sizes, int n_in,
                              void* d_out, int out_size, void* d_ws, size_t ws_size,
                              hipStream_t stream) {
  (void)in_sizes; (void)n_in; (void)out_size; (void)ws_size;
  const float* x  = (const float*)d_in[0];
  const float* Wq = (const float*)d_in[1];
  const float* Wk = (const float*)d_in[2];
  const float* Wv = (const float*)d_in[3];
  const float* Wo = (const float*)d_in[4];
  float* out = (float*)d_out;

  const size_t R   = RTOT;                // 4096 rows
  const size_t KVW = (size_t)NKV * HDIM;  // 512

  char* w = (char*)d_ws;
  __bf16* xb   = (__bf16*)w; w += R * CDIM * 2;
  __bf16* WqT  = (__bf16*)w; w += (size_t)CDIM * CDIM * 2;   // [N=2048][K=2048]
  __bf16* WkT  = (__bf16*)w; w += KVW * CDIM * 2;            // [N=512][K=2048]
  __bf16* WvT  = (__bf16*)w; w += KVW * CDIM * 2;
  __bf16* WoT  = (__bf16*)w; w += (size_t)CDIM * CDIM * 2;
  __bf16* qbuf = (__bf16*)w; w += R * CDIM * 2;
  __bf16* kbuf = (__bf16*)w; w += R * KVW * 2;
  __bf16* vtbuf= (__bf16*)w; w += KVW * R * 2;               // transposed V [512][4096]
  __bf16* ybuf = (__bf16*)w; w += R * CDIM * 2;

  // x: plain convert (A-side of GEMMs). Weights: convert + transpose to [N][K].
  cvt_f32_bf16<<<dim3((unsigned)((R * CDIM + 255) / 256)), 256, 0, stream>>>(
      x, xb, (int)(R * CDIM));
  cvt_t_f32_bf16<<<dim3((unsigned)(((size_t)CDIM * CDIM + 255) / 256)), 256, 0, stream>>>(
      Wq, WqT, CDIM, CDIM);
  cvt_t_f32_bf16<<<dim3((unsigned)((CDIM * KVW + 255) / 256)), 256, 0, stream>>>(
      Wk, WkT, CDIM, (int)KVW);
  cvt_t_f32_bf16<<<dim3((unsigned)((CDIM * KVW + 255) / 256)), 256, 0, stream>>>(
      Wv, WvT, CDIM, (int)KVW);
  cvt_t_f32_bf16<<<dim3((unsigned)(((size_t)CDIM * CDIM + 255) / 256)), 256, 0, stream>>>(
      Wo, WoT, CDIM, CDIM);

  // projections (all-async staging)
  gemm_bf16_wmma<__bf16, false><<<dim3(CDIM / 128, (unsigned)(R / 128)), 256, 0, stream>>>(
      xb, WqT, qbuf, (int)R, CDIM, CDIM);
  gemm_bf16_wmma<__bf16, false><<<dim3((unsigned)(KVW / 128), (unsigned)(R / 128)), 256, 0, stream>>>(
      xb, WkT, kbuf, (int)R, (int)KVW, CDIM);
  gemm_bf16_wmma<__bf16, true><<<dim3((unsigned)(KVW / 128), (unsigned)(R / 128)), 256, 0, stream>>>(
      xb, WvT, vtbuf, (int)R, (int)KVW, CDIM);   // writes vT[col][row]

  // RoPE on q and k
  {
    size_t nq = R * NHEAD * (HDIM / 2);
    rope_bf16<<<dim3((unsigned)((nq + 255) / 256)), 256, 0, stream>>>(qbuf, (int)R, NHEAD, CDIM);
    size_t nk = R * NKV * (HDIM / 2);
    rope_bf16<<<dim3((unsigned)((nk + 255) / 256)), 256, 0, stream>>>(kbuf, (int)R, NKV, (int)KVW);
  }

  // attention
  flash_attn_bf16<<<dim3(TSEQ / 64, NHEAD, 2), 128, 0, stream>>>(qbuf, kbuf, vtbuf, ybuf);

  // output projection (f32 out)
  gemm_bf16_wmma<float, false><<<dim3(CDIM / 128, (unsigned)(R / 128)), 256, 0, stream>>>(
      ybuf, WoT, out, (int)R, CDIM, CDIM);
}
